// PointMLP_28243704939067
// MI455X (gfx1250) — matrix operand order repaired and checked by reading, not compiled, not run on hardware
//
#include <hip/hip_runtime.h>

typedef __attribute__((ext_vector_type(16))) __bf16 v16bf;
typedef __attribute__((ext_vector_type(8)))  __bf16 v8bf;
typedef __attribute__((ext_vector_type(8)))  float  v8f;

#define EPS 1e-5f
#define MT 4   // 4 WMMA M-tiles per wave: 64x16 output per wave32

// ------------------------------------------------------------------
// Fused GEMM + folded-BN (+residual)(+ReLU) using v_wmma_f32_16x16x32_bf16.
// All WMMA operands are pre-converted bf16 (weights converted once per
// launch; activations produced in bf16 by the previous kernel), so the hot
// loop has NO f32->bf16 conversion ALU: just 10 contiguous b128 loads and
// 4 WMMAs per K-step (B fragment reused across 4 M-tiles).
//   W: Cout x Cin bf16 row-major.  X: G x L x Cin bf16 (channel-contiguous).
//   Yf (f32, optional) / Yh (bf16): G x L x Cout.   R (f32 residual) optional.
//   Cin % 32 == 0, Cout % 64 == 0.
// Padded columns (n >= L): loads clamp to a valid row (WMMA columns are
// independent), stores are skipped -> no exec-mask churn in the loop.
// ------------------------------------------------------------------
__global__ __launch_bounds__(32) void gemm_bn_kernel(
    const __bf16* __restrict__ W, const __bf16* __restrict__ X,
    const float* __restrict__ scale, const float* __restrict__ shift,
    const float* __restrict__ R, float* __restrict__ Yf, __bf16* __restrict__ Yh,
    int Cin, int Cout, int L, int relu)
{
  int ntil  = (L + 15) >> 4;
  int mtil  = Cout >> 6;            // 64-row super-tiles
  int per_g = ntil * mtil;
  int bid = blockIdx.x;
  int g  = bid / per_g;
  int t  = bid - g * per_g;
  int mt = t / ntil;
  int nt = t - mt * ntil;

  int lane = threadIdx.x;           // wave32
  int half = lane >> 4;             // 0: lanes 0-15, 1: lanes 16-31
  int l16  = lane & 15;

  int n  = nt * 16 + l16;           // output column (spatial position)
  int nc = (n < L) ? n : (L - 1);   // clamped row for loads
  const __bf16* Xrow = X + ((size_t)g * L + nc) * Cin;

  const __bf16* Wr[MT];
#pragma unroll
  for (int mi = 0; mi < MT; ++mi)
    Wr[mi] = W + (size_t)((mt * MT + mi) * 16 + l16) * Cin;   // A row M=l16

  v8f acc[MT];
#pragma unroll
  for (int mi = 0; mi < MT; ++mi) acc[mi] = (v8f){0.f,0.f,0.f,0.f,0.f,0.f,0.f,0.f};

  for (int k0 = 0; k0 < Cin; k0 += 32) {
    // B fragment: lane holds col n, slots 0..7 -> K=kb..kb+7, slots 8..15 -> K=kb+16..kb+23
    int kb = k0 + (half ? 8 : 0);
    v8bf b0 = *(const v8bf*)(Xrow + kb);
    v8bf b1 = *(const v8bf*)(Xrow + kb + 16);
    v16bf b = __builtin_shufflevector(b0, b1, 0,1,2,3,4,5,6,7,8,9,10,11,12,13,14,15);

    // A fragments: slots 0..7 -> K=ka..ka+7, slots 8..15 -> K=ka+16..ka+23
    int ka = k0 + (half ? 8 : 0);
#pragma unroll
    for (int mi = 0; mi < MT; ++mi) {
      v8bf a0 = *(const v8bf*)(Wr[mi] + ka);
      v8bf a1 = *(const v8bf*)(Wr[mi] + ka + 16);
      v16bf a = __builtin_shufflevector(a0, a1, 0,1,2,3,4,5,6,7,8,9,10,11,12,13,14,15);
      acc[mi] = __builtin_amdgcn_wmma_f32_16x16x32_bf16(false, a, false, b,
                                                        (short)0, acc[mi], false, false);
    }
  }

  if (n < L) {
    float*  Yfrow = Yf ? Yf + ((size_t)g * L + n) * Cout : nullptr;
    __bf16* Yhrow = Yh + ((size_t)g * L + n) * Cout;
    const float* Rrow = R ? R + ((size_t)g * L + n) * Cout : nullptr;
#pragma unroll
    for (int mi = 0; mi < MT; ++mi) {
      int m0 = (mt * MT + mi) * 16 + 8 * half;   // D: VGPR r -> row m0 + r
      float4 sc0 = *(const float4*)(scale + m0);
      float4 sc1 = *(const float4*)(scale + m0 + 4);
      float4 sf0 = *(const float4*)(shift + m0);
      float4 sf1 = *(const float4*)(shift + m0 + 4);
      float o[8];
      o[0]=sc0.x*acc[mi][0]+sf0.x; o[1]=sc0.y*acc[mi][1]+sf0.y;
      o[2]=sc0.z*acc[mi][2]+sf0.z; o[3]=sc0.w*acc[mi][3]+sf0.w;
      o[4]=sc1.x*acc[mi][4]+sf1.x; o[5]=sc1.y*acc[mi][5]+sf1.y;
      o[6]=sc1.z*acc[mi][6]+sf1.z; o[7]=sc1.w*acc[mi][7]+sf1.w;
      if (Rrow) {
        float4 r0 = *(const float4*)(Rrow + m0);
        float4 r1 = *(const float4*)(Rrow + m0 + 4);
        o[0]+=r0.x; o[1]+=r0.y; o[2]+=r0.z; o[3]+=r0.w;
        o[4]+=r1.x; o[5]+=r1.y; o[6]+=r1.z; o[7]+=r1.w;
      }
      if (relu) {
#pragma unroll
        for (int r = 0; r < 8; ++r) o[r] = fmaxf(o[r], 0.f);
      }
      if (Yfrow) {
        *(float4*)(Yfrow + m0)     = make_float4(o[0], o[1], o[2], o[3]);
        *(float4*)(Yfrow + m0 + 4) = make_float4(o[4], o[5], o[6], o[7]);
      }
      v8bf oh;
#pragma unroll
      for (int r = 0; r < 8; ++r) oh[r] = (__bf16)o[r];   // native v_cvt path
      *(v8bf*)(Yhrow + m0) = oh;
    }
  }
}

// f32 -> bf16 weight pre-conversion (once per launch; weights are hot-reused)
__global__ void cvt_bf16_kernel(const float* __restrict__ W, __bf16* __restrict__ Wh, int n)
{
  int i = blockIdx.x * blockDim.x + threadIdx.x;
  if (i < n) Wh[i] = (__bf16)W[i];
}

// BN folding: scale = g/sqrt(v+eps), shift = (b-m)*scale + bt
__global__ void bn_fold_kernel(const float* __restrict__ b, const float* __restrict__ g,
                               const float* __restrict__ bt, const float* __restrict__ m,
                               const float* __restrict__ v, float* __restrict__ scale,
                               float* __restrict__ shift, int C)
{
  int i = blockIdx.x * blockDim.x + threadIdx.x;
  if (i >= C) return;
  float s = g[i] * rsqrtf(v[i] + EPS);
  scale[i] = s;
  shift[i] = (b[i] - m[i]) * s + bt[i];
}

// Embedding: feats[b][n][c] (channel-last f32) = relu(bn(W(64x3) @ x[b][n]))
__global__ void embed_kernel(const float* __restrict__ x, const float* __restrict__ w,
                             const float* __restrict__ scale, const float* __restrict__ shift,
                             float* __restrict__ out, int total)
{
  int i = blockIdx.x * blockDim.x + threadIdx.x;
  if (i >= total) return;
  int c = i & 63;
  int bn = i >> 6;
  const float* p = x + (size_t)bn * 3;
  float v = w[c*3+0]*p[0] + w[c*3+1]*p[1] + w[c*3+2]*p[2];
  v = scale[c] * v + shift[c];
  out[i] = fmaxf(v, 0.f);
}

// Farthest point sampling, one 256-thread block per batch. Matches the scan
// semantics: records incoming `far` (first = 0), argmax tie -> lowest index.
__global__ void fps_kernel(const float* __restrict__ pts, int* __restrict__ idx, int N, int num)
{
  __shared__ float mind[1024];
  __shared__ float rd[256];
  __shared__ int   ri[256];
  __shared__ int   s_far;
  int b = blockIdx.x;
  int t = threadIdx.x;
  const float* P = pts + (size_t)b * N * 3;
  for (int n = t; n < N; n += 256) mind[n] = 1e10f;
  if (t == 0) s_far = 0;
  __syncthreads();
  for (int it = 0; it < num; ++it) {
    int far = s_far;
    if (t == 0) idx[b * num + it] = far;
    float fx = P[far*3], fy = P[far*3+1], fz = P[far*3+2];
    float bd = -1.f; int bi = 0x7fffffff;
    for (int n = t; n < N; n += 256) {
      float dx = P[n*3]-fx, dy = P[n*3+1]-fy, dz = P[n*3+2]-fz;
      float d = dx*dx + dy*dy + dz*dz;
      float md = fminf(mind[n], d);
      mind[n] = md;
      if (md > bd) { bd = md; bi = n; }
    }
    rd[t] = bd; ri[t] = bi;
    __syncthreads();
    for (int h = 128; h > 0; h >>= 1) {
      if (t < h) {
        if (rd[t+h] > rd[t] || (rd[t+h] == rd[t] && ri[t+h] < ri[t])) {
          rd[t] = rd[t+h]; ri[t] = ri[t+h];
        }
      }
      __syncthreads();
    }
    if (t == 0) s_far = ri[0];
    __syncthreads();
  }
}

// complement_idx: first N-S indices not in fps_idx, ascending
__global__ void complement_kernel(const int* __restrict__ fi, int S, int* __restrict__ out, int N)
{
  __shared__ unsigned char mask[1024];
  int b = blockIdx.x;
  for (int n = threadIdx.x; n < N; n += blockDim.x) mask[n] = 0;
  __syncthreads();
  for (int s = threadIdx.x; s < S; s += blockDim.x) mask[fi[b*S+s]] = 1;
  __syncthreads();
  if (threadIdx.x == 0) {
    int c = 0, M = N - S;
    for (int n = 0; n < N && c < M; ++n)
      if (!mask[n]) out[b*M + (c++)] = n;
  }
}

// Geometric affine pass 1: per (b,s) select 24 nearest neighbors, write new
// center points, neighbor indices, the "mean" half (channels C..2C-1) of the
// bf16 group matrix Xh (G x 24 x 2C channel-last), a f32 means side-buffer,
// and per-group partial sum/sumsq for the deterministic batch std.
__global__ void ga_topk_kernel(
    const float* __restrict__ pts, int N,
    const float* __restrict__ feats, long long sB, long long sN, int C,
    const int* __restrict__ sidx, int S,
    float* __restrict__ newpts, int* __restrict__ gidx,
    __bf16* __restrict__ Xh, float* __restrict__ means, float* __restrict__ partial)
{
  __shared__ float dist[1024];
  __shared__ float rd[128];
  __shared__ int   ri[128];
  __shared__ int   sel[24];
  int bs = blockIdx.x;
  int b = bs / S;
  int t = threadIdx.x;
  const float* P = pts + (size_t)b * N * 3;
  int si = sidx[bs];
  float sx = P[si*3], sy = P[si*3+1], sz = P[si*3+2];
  if (t < 3) newpts[bs*3 + t] = P[si*3 + t];
  for (int n = t; n < N; n += 128) {
    float dx = P[n*3]-sx, dy = P[n*3+1]-sy, dz = P[n*3+2]-sz;
    dist[n] = dx*dx + dy*dy + dz*dz;
  }
  __syncthreads();
  for (int j = 0; j < 24; ++j) {
    float bd = 1e30f; int bi = 0x7fffffff;
    for (int n = t; n < N; n += 128) {
      float d = dist[n];
      if (d < bd) { bd = d; bi = n; }
    }
    rd[t] = bd; ri[t] = bi;
    __syncthreads();
    for (int h = 64; h > 0; h >>= 1) {
      if (t < h) {
        if (rd[t+h] < rd[t] || (rd[t+h] == rd[t] && ri[t+h] < ri[t])) {
          rd[t] = rd[t+h]; ri[t] = ri[t+h];
        }
      }
      __syncthreads();
    }
    if (t == 0) { sel[j] = ri[0]; dist[ri[0]] = 1e30f; gidx[(size_t)bs*24 + j] = ri[0]; }
    __syncthreads();
  }
  int C2 = 2 * C;
  const float* fb = feats + (size_t)b * sB;
  __bf16* Xg = Xh + (size_t)bs * 24 * C2;
  float lsum = 0.f, lsq = 0.f;
  for (int c = t; c < C; c += 128) {
    float mean = fb[(size_t)si * sN + c];
    means[(size_t)bs * C + c] = mean;
    __bf16 mh = (__bf16)mean;
#pragma unroll
    for (int j = 0; j < 24; ++j) {
      Xg[(size_t)j * C2 + C + c] = mh;
      float y = fb[(size_t)sel[j] * sN + c] - mean;
      lsum += y; lsq += y * y;
    }
  }
  rd[t] = lsum;
  __syncthreads();
  for (int h = 64; h > 0; h >>= 1) { if (t < h) rd[t] += rd[t+h]; __syncthreads(); }
  if (t == 0) partial[bs*2 + 0] = rd[0];
  __syncthreads();
  rd[t] = lsq;
  __syncthreads();
  for (int h = 64; h > 0; h >>= 1) { if (t < h) rd[t] += rd[t+h]; __syncthreads(); }
  if (t == 0) partial[bs*2 + 1] = rd[0];
}

// Batch-wide std with ddof=1 over S*24*C values (deterministic sequential sum).
__global__ void ga_std_kernel(const float* __restrict__ partial, float* __restrict__ stdb,
                              int B, int S, int C)
{
  int b = blockIdx.x * blockDim.x + threadIdx.x;
  if (b >= B) return;
  float sum = 0.f, sq = 0.f;
  for (int s = 0; s < S; ++s) { sum += partial[(b*S+s)*2]; sq += partial[(b*S+s)*2+1]; }
  float n = (float)S * 24.f * (float)C;
  float var = (sq - sum * sum / n) / (n - 1.f);
  stdb[b] = sqrtf(fmaxf(var, 0.f));
}

// Geometric affine pass 2: Xh[g][j][c] = bf16(alpha*((gf-mean)/(std+eps)) + beta)
__global__ void ga_norm_kernel(
    const float* __restrict__ feats, long long sB, long long sN, int C,
    const int* __restrict__ gidx, const float* __restrict__ stdb,
    const float* __restrict__ means,
    const float* __restrict__ alpha, const float* __restrict__ beta,
    __bf16* __restrict__ Xh, int S, int total)
{
  int i = blockIdx.x * blockDim.x + threadIdx.x;
  if (i >= total) return;
  int bs = i / C;
  int c = i - bs * C;
  int b = bs / S;
  int C2 = 2 * C;
  __bf16* Xg = Xh + (size_t)bs * 24 * C2;
  float mean = means[(size_t)bs * C + c];
  float inv = 1.f / (stdb[b] + EPS);
  float al = alpha[c], be = beta[c];
  const float* fb = feats + (size_t)b * sB;
#pragma unroll
  for (int j = 0; j < 24; ++j) {
    int n = gidx[(size_t)bs*24 + j];
    float y = (fb[(size_t)n * sN + c] - mean) * inv;
    Xg[(size_t)j * C2 + c] = (__bf16)(al * y + be);
  }
}

// max over k=24 neighbors: f/fh[(b*S+s)*C + c] = max_j Yf[((b*S+s)*24 + j)*C + c]
__global__ void maxpool_k_kernel(const float* __restrict__ Yf, float* __restrict__ f,
                                 __bf16* __restrict__ fh, int C, int total)
{
  int i = blockIdx.x * blockDim.x + threadIdx.x;
  if (i >= total) return;
  int bs = i / C;
  int c = i - bs * C;
  const float* Yg = Yf + (size_t)bs * 24 * C + c;
  float m = Yg[0];
#pragma unroll
  for (int j = 1; j < 24; ++j) m = fmaxf(m, Yg[(size_t)j * C]);
  f[i] = m;
  fh[i] = (__bf16)m;
}

// max over S: g[b*C + c] = max_s f[(b*S+s)*C + c]   (f is B x S x C f32)
__global__ void maxpool_s_kernel(const float* __restrict__ f, float* __restrict__ g,
                                 int C, int S, int total)
{
  int i = blockIdx.x * blockDim.x + threadIdx.x;
  if (i >= total) return;
  int b = i / C;
  int c = i - b * C;
  const float* fr = f + (size_t)b * S * C + c;
  float m = fr[0];
  for (int s = 1; s < S; ++s) m = fmaxf(m, fr[(size_t)s * C]);
  g[i] = m;
}

// Classifier linear (+ folded bn1d)(+relu). scale==nullptr -> plain bias in shift.
__global__ void linear_kernel(const float* __restrict__ X, int Cin,
                              const float* __restrict__ W, int Cout,
                              const float* __restrict__ scale, const float* __restrict__ shift,
                              float* __restrict__ Y, int relu, int total)
{
  int i = blockIdx.x * blockDim.x + threadIdx.x;
  if (i >= total) return;
  int b = i / Cout, o = i - b * Cout;
  const float* xr = X + (size_t)b * Cin;
  const float* wr = W + (size_t)o * Cin;
  float acc = 0.f;
  for (int c = 0; c < Cin; ++c) acc += xr[c] * wr[c];
  float v = scale ? (scale[o] * acc + shift[o]) : (acc + shift[o]);
  if (relu) v = fmaxf(v, 0.f);
  Y[i] = v;
}

struct Fold { float* scale; float* shift; };
struct CBp  { const float *b,*bt,*g,*m,*v,*w; };

extern "C" void kernel_launch(void* const* d_in, const int* in_sizes, int n_in,
                              void* d_out, int out_size, void* d_ws, size_t ws_size,
                              hipStream_t stream) {
  (void)in_sizes; (void)n_in; (void)out_size; (void)ws_size;
  const int B = 8, N0 = 1024;
  const int ga_ch[4]    = {64, 128, 256, 512};
  const int out_ch[4]   = {128, 256, 512, 1024};
  const int fps_nums[4] = {512, 256, 128, 64};

  // ---- unpack parameters: JAX pytree order (dict keys sorted) ----
  int cur = 0;
  auto F = [&]() { return (const float*)d_in[cur++]; };
  auto readCB = [&]() { CBp c; c.b=F(); c.bt=F(); c.g=F(); c.m=F(); c.v=F(); c.w=F(); return c; };
  const float *bn1_bt=F(), *bn1_g=F(), *bn1_m=F(), *bn1_v=F();
  const float *bn2_bt=F(), *bn2_g=F(), *bn2_m=F(), *bn2_v=F();
  const float *l1_b=F(), *l1_w=F();
  const float *l2_b=F(), *l2_w=F();
  const float *l3_b=F(), *l3_w=F();
  CBp emb = readCB();
  const float *gaA[4], *gaB[4];
  for (int i=0;i<4;++i){ gaA[i]=F(); gaB[i]=F(); }
  CBp pos_c[4][4];
  for (int i=0;i<4;++i) for (int j=0;j<4;++j) pos_c[i][j]=readCB();
  CBp pre_b0[4]; CBp pre_c[4][4];
  for (int i=0;i<4;++i){ pre_b0[i]=readCB(); for (int j=0;j<4;++j) pre_c[i][j]=readCB(); }
  const float* x = F();

  // ---- workspace bump allocator (same offsets every call: deterministic) ----
  float* wsf = (float*)d_ws;
  size_t off = 0;
  auto alloc  = [&](size_t n){ float* p = wsf + off; off += n; return p; };
  auto allocH = [&](size_t n){ return (__bf16*)alloc((n + 1) / 2); };  // n bf16

  float* feats0 = alloc((size_t)B * N0 * 64);       // (B,N,64) channel-last f32
  const size_t TILE = 524288ull * 24ull;            // G*24*C is constant per stage
  __bf16* Xh  = allocH(TILE);                       // (G, 24, 2C) bf16
  float*  Ya  = alloc(TILE);                        // f32 Y buffers (residual sources)
  float*  Yb  = alloc(TILE);
  float*  Yc  = alloc(TILE);
  __bf16* Yah = allocH(TILE);                       // bf16 twins (next-GEMM inputs)
  __bf16* Ybh = allocH(TILE);
  __bf16* Ych = allocH(TILE);
  int*   gidx = (int*)alloc(4096 * 24);
  float* partial = alloc(4096 * 2);
  float* means = alloc(262144);                     // max G*C
  float* stdb = alloc(8);
  int* fiBuf  = (int*)alloc(4096);
  int* fi2Buf = (int*)alloc(4096);
  float *ptsA[2], *ptsB[2], *fA[2], *fB[2];
  __bf16 *fAh[2], *fBh[2];
  for (int br=0;br<2;++br){ ptsA[br]=alloc((size_t)B*512*3); ptsB[br]=alloc((size_t)B*512*3); }
  for (int br=0;br<2;++br){ fA[br]=alloc(524288); fB[br]=alloc(524288);
                            fAh[br]=allocH(524288); fBh[br]=allocH(524288); }
  float* h1 = alloc((size_t)B*512);
  float* h2 = alloc((size_t)B*256);

  auto fold = [&](const CBp& c, int C){
    Fold f; f.scale = alloc(C); f.shift = alloc(C);
    bn_fold_kernel<<<(C+127)/128, 128, 0, stream>>>(c.b, c.g, c.bt, c.m, c.v, f.scale, f.shift, C);
    return f;
  };
  auto cvtW = [&](const float* W, int n){
    __bf16* Wh = allocH((size_t)n);
    cvt_bf16_kernel<<<(n+255)/256, 256, 0, stream>>>(W, Wh, n);
    return Wh;
  };

  Fold f_emb = fold(emb, 64);
  Fold f_pre_b0[4], f_pre_c[4][4], f_pos_c[4][4];
  __bf16 *Wh_pre_b0[4], *Wh_pre_c[4][4], *Wh_pos_c[4][4];
  for (int i=0;i<4;++i){
    int Co = out_ch[i], C2 = 2*ga_ch[i];
    f_pre_b0[i]  = fold(pre_b0[i], Co);
    Wh_pre_b0[i] = cvtW(pre_b0[i].w, Co * C2);
    for (int j=0;j<4;++j){
      f_pre_c[i][j]  = fold(pre_c[i][j], Co);
      f_pos_c[i][j]  = fold(pos_c[i][j], Co);
      Wh_pre_c[i][j] = cvtW(pre_c[i][j].w, Co * Co);
      Wh_pos_c[i][j] = cvtW(pos_c[i][j].w, Co * Co);
    }
  }
  float *s1 = alloc(512), *sh1 = alloc(512);
  bn_fold_kernel<<<4, 128, 0, stream>>>(l1_b, bn1_g, bn1_bt, bn1_m, bn1_v, s1, sh1, 512);
  float *s2 = alloc(256), *sh2 = alloc(256);
  bn_fold_kernel<<<2, 128, 0, stream>>>(l2_b, bn2_g, bn2_bt, bn2_m, bn2_v, s2, sh2, 256);

  auto gemm = [&](const __bf16* W, const __bf16* Xin, const Fold& fo, const float* R,
                  float* Yf, __bf16* Yh, int Cin, int Cout, int L, int G, int relu){
    int blocks = G * (Cout/64) * ((L+15)/16);
    gemm_bn_kernel<<<blocks, 32, 0, stream>>>(W, Xin, fo.scale, fo.shift, R, Yf, Yh,
                                              Cin, Cout, L, relu);
  };
  auto run_ga = [&](const float* pts, int N, const float* feats,
                    long long sB, long long sN, int C,
                    const int* sidx, int S, float* newpts, int st){
    ga_topk_kernel<<<B*S, 128, 0, stream>>>(pts, N, feats, sB, sN, C, sidx, S,
                                            newpts, gidx, Xh, means, partial);
    ga_std_kernel<<<1, 32, 0, stream>>>(partial, stdb, B, S, C);
    int total = B * S * C;
    ga_norm_kernel<<<(total+255)/256, 256, 0, stream>>>(feats, sB, sN, C, gidx, stdb,
                                                        means, gaA[st], gaB[st], Xh, S, total);
  };
  auto run_pre = [&](int i, int S, float* fout, __bf16* fouth){
    int C2 = 2*ga_ch[i], Co = out_ch[i], G = B*S;
    gemm(Wh_pre_b0[i],  Xh,  f_pre_b0[i],  nullptr, Ya, Yah, C2, Co, 24, G, 1);  // cbr b0
    gemm(Wh_pre_c[i][0], Yah, f_pre_c[i][0], nullptr, nullptr, Ybh, Co, Co, 24, G, 1); // r1.c1
    gemm(Wh_pre_c[i][1], Ybh, f_pre_c[i][1], Ya,     Yc, Ych, Co, Co, 24, G, 1); // r1.c2 + res
    gemm(Wh_pre_c[i][2], Ych, f_pre_c[i][2], nullptr, nullptr, Yah, Co, Co, 24, G, 1); // r2.c1
    gemm(Wh_pre_c[i][3], Yah, f_pre_c[i][3], Yc,     Yb, Ybh, Co, Co, 24, G, 1); // r2.c2 + res
    int total = B * S * Co;
    maxpool_k_kernel<<<(total+255)/256, 256, 0, stream>>>(Yb, fout, fouth, Co, total);
  };
  auto run_pos = [&](int i, int S, float* fa, __bf16* fah, float* fb, __bf16* fbh){
    int Co = out_ch[i];
    gemm(Wh_pos_c[i][0], fah, f_pos_c[i][0], nullptr, nullptr, Yah, Co, Co, S, B, 1);
    gemm(Wh_pos_c[i][1], Yah, f_pos_c[i][1], fa,      fb, fbh, Co, Co, S, B, 1);
    gemm(Wh_pos_c[i][2], fbh, f_pos_c[i][2], nullptr, nullptr, Yah, Co, Co, S, B, 1);
    gemm(Wh_pos_c[i][3], Yah, f_pos_c[i][3], fb,      fa, fah, Co, Co, S, B, 1); // result: fa/fah
  };

  // ---- embedding ----
  { int total = B*N0*64;
    embed_kernel<<<(total+255)/256, 256, 0, stream>>>(x, emb.w, f_emb.scale, f_emb.shift, feats0, total); }

  // ---- stage 0, branch 1 (FPS half) and branch 2 (complement half) ----
  fps_kernel<<<B, 256, 0, stream>>>(x, fiBuf, N0, 512);
  run_ga(x, N0, feats0, (long long)N0*64, 64, 64, fiBuf, 512, ptsA[0], 0);
  run_pre(0, 512, fA[0], fAh[0]);
  run_pos(0, 512, fA[0], fAh[0], fB[0], fBh[0]);

  complement_kernel<<<B, 256, 0, stream>>>(fiBuf, 512, fi2Buf, N0);
  run_ga(x, N0, feats0, (long long)N0*64, 64, 64, fi2Buf, 512, ptsA[1], 0);
  run_pre(0, 512, fA[1], fAh[1]);
  run_pos(0, 512, fA[1], fAh[1], fB[1], fBh[1]);

  // ---- stages 1..3 ----
  int Sprev = 512;
  for (int i = 1; i < 4; ++i) {
    int S = fps_nums[i];
    int Cf = out_ch[i-1];                  // == ga_ch[i]
    for (int br = 0; br < 2; ++br) {
      fps_kernel<<<B, 256, 0, stream>>>(ptsA[br], fiBuf, Sprev, S);
      // feats = f (B,Sprev,Cf) channel-last -> sB = Sprev*Cf, sN = Cf
      run_ga(ptsA[br], Sprev, fA[br], (long long)Sprev*Cf, Cf, Cf, fiBuf, S, ptsB[br], i);
      run_pre(i, S, fA[br], fAh[br]);
      run_pos(i, S, fA[br], fAh[br], fB[br], fBh[br]);
      float* tp = ptsA[br]; ptsA[br] = ptsB[br]; ptsB[br] = tp;
    }
    Sprev = S;
  }

  // ---- global pooling + classifier; d_out = [logits(8x15), g1(8x1024), g2(8x1024)] ----
  float* outF = (float*)d_out;
  float* g1 = outF + 120;
  float* g2 = outF + 120 + B*1024;
  { int total = B*1024;
    maxpool_s_kernel<<<(total+255)/256, 256, 0, stream>>>(fA[0], g1, 1024, 64, total);
    maxpool_s_kernel<<<(total+255)/256, 256, 0, stream>>>(fA[1], g2, 1024, 64, total); }
  { int t1 = B*512;
    linear_kernel<<<(t1+127)/128, 128, 0, stream>>>(g1, 1024, l1_w, 512, s1, sh1, h1, 1, t1);
    int t2 = B*256;
    linear_kernel<<<(t2+127)/128, 128, 0, stream>>>(h1, 512, l2_w, 256, s2, sh2, h2, 1, t2);
    int t3 = B*15;
    linear_kernel<<<(t3+127)/128, 128, 0, stream>>>(h2, 256, l3_w, 15, nullptr, l3_b, outF, 0, t3); }
}
